// EANet_18820546691345
// MI455X (gfx1250) — compile-verified, matrix operand order checked by reading, not run on hardware
//
#include <hip/hip_runtime.h>
#include <hip/hip_bf16.h>
#include <math.h>

typedef _Float16 half_t;
typedef __attribute__((ext_vector_type(16))) _Float16 v16h;
typedef __attribute__((ext_vector_type(8)))  _Float16 v8h;
typedef __attribute__((ext_vector_type(8)))  float    v8f;

#define NB      128
#define NE      64
#define NUPE    32
#define NA      16
#define DIMF    256
#define F12F    768
#define NRBF    16
#define NLAY    4

#define TANH_GAIN 1.5926535f
#define SILU_GAIN 1.7881294f
#define RSQRT2    0.70710678118654752f

// ---------------- workspace layout (bytes) ----------------
#define WS_MPW   0                               // 16*16 f32
#define WS_SPD   1024                            // 16 f32 (softplus(sigma))
#define WS_WATOM 2048                            // 1536*256 halves, frag-packed
#define WS_WLIN  (WS_WATOM + 1536*256*2)         // 768*256 halves
#define WS_WSELF (WS_WLIN  + 768*256*2)          // 4*256*256 halves
#define WS_WMSG  (WS_WSELF + 4*256*256*2)        // 4*256*256 halves

// ---------------- LDS layout (bytes), all 16B aligned ----------------
#define L_ELEC    0        // 64*3 f32   (768)
#define L_ATOMS   768      // 16*3 f32   (192)
#define L_NUC     960      // 16*4 f32   (256)
#define L_MU      1216     // 16 f32     (64)
#define L_SPD     1280     // 16 f32     (64)
#define L_RBAR    2048     // 64*16 f32  (4096)
#define L_RBMEAN  6144     // 64 f32     (256)
#define L_R4      6400     // 64*16*4 f32 (16384)
#define L_H2CAT   22784    // 64*512 f16 (65536)   -- dead after phase 2b
// phase-3 overlays inside the H2CAT region:
#define L_AEMB    22784    // 16*256 f32 (16384)
#define L_PRE     39168    // 16*256 f32 (16384)
#define L_AEMBH   55552    // 16*256 f16 (8192)
#define L_MSGH    63744    // 16*256 f16 (8192)
#define L_MPW     71936    // 16*16  f32 (1024)
#define L_ELECA   88320    // 64*768 f16 (98304)
#define L_ATOMCAT 186624   // 16*1536 f16 (49152)
#define L_TOTAL   235776

__device__ __forceinline__ float silu_g(float x) {
  return (x / (1.0f + __expf(-x))) * SILU_GAIN;
}

__device__ __forceinline__ v8f wmma16(v16h a, v16h b, v8f c) {
  // D = A(16x32 f16) * B(32x16 f16) + C(16x16 f32)
  return __builtin_amdgcn_wmma_f32_16x16x32_f16(false, a, false, b, (short)0, c,
                                                false, false);
}

// A fragment from an LDS row-major [M][Kstride] f16 matrix.
// 16-bit A 16x32 layout: lane M = lane&15; lanes>=16 use K-base +8;
// frag element v maps to K = kbase + (v<8 ? v : v+8).
__device__ __forceinline__ v16h load_a_frag(const half_t* row, int kbase, int lane) {
  const half_t* p = row + kbase + ((lane >> 4) & 1) * 8;
  v8h lo = *(const v8h*)(p);
  v8h hi = *(const v8h*)(p + 16);
  v16h a;
#pragma unroll
  for (int t = 0; t < 8; ++t) { a[t] = lo[t]; a[t + 8] = hi[t]; }
  return a;
}

// B fragment from frag-packed global weights: contiguous 32B per (k0,n,half).
__device__ __forceinline__ v16h load_b_frag(const half_t* packed, int k0, int n, int lane) {
  const v16h* p = (const v16h*)packed + (((size_t)k0 * 256 + n) * 2 + ((lane >> 4) & 1));
  return *p;
}

// ------------------------------------------------------------------
// prep: mp_weight softmax + softplus(sigma)
// ------------------------------------------------------------------
__global__ void eanet_prep_small(const float* __restrict__ atoms,
                                 const float* __restrict__ temperature,
                                 const float* __restrict__ gsigma,
                                 float* __restrict__ ws_mpw,
                                 float* __restrict__ ws_spd) {
  int t = threadIdx.x;
  if (t < NA) {
    int m = t;
    float ax = atoms[m * 3 + 0], ay = atoms[m * 3 + 1], az = atoms[m * 3 + 2];
    float tm = temperature[m];
    float d[NA];
    float mx = -1e30f;
#pragma unroll
    for (int n = 0; n < NA; ++n) {
      float dx = atoms[n * 3 + 0] - ax;
      float dy = atoms[n * 3 + 1] - ay;
      float dz = atoms[n * 3 + 2] - az;
      float v = -sqrtf(dx * dx + dy * dy + dz * dz) * tm;
      if (n == m) v -= 10000.0f;
      d[n] = v;
      mx = fmaxf(mx, v);
    }
    float s = 0.0f;
#pragma unroll
    for (int n = 0; n < NA; ++n) { float e = __expf(d[n] - mx); d[n] = e; s += e; }
    float inv = 1.0f / s;
#pragma unroll
    for (int n = 0; n < NA; ++n) ws_mpw[m * NA + n] = d[n] * inv;
  } else if (t < 2 * NA) {
    int r = t - NA;
    float x = gsigma[r];
    ws_spd[r] = (x > 20.0f) ? x : log1pf(__expf(x));
  }
}

// ------------------------------------------------------------------
// prep: convert an (K x 256) f32 weight to f16 WMMA-B fragment order.
// packed idx = (((k0*256)+n)*2 + h)*16 + v ;  k = k0*32 + h*8 + (v<8?v:v+8)
// ------------------------------------------------------------------
__global__ void eanet_pack_w(const float* __restrict__ src,
                             half_t* __restrict__ dst, int total) {
  int idx = blockIdx.x * blockDim.x + threadIdx.x;
  if (idx >= total) return;
  int v  = idx & 15;
  int h  = (idx >> 4) & 1;
  int n  = (idx >> 5) & 255;
  int k0 = idx >> 13;
  int k  = k0 * 32 + h * 8 + (v < 8 ? v : v + 8);
  dst[idx] = (half_t)src[(size_t)k * 256 + n];
}

// ------------------------------------------------------------------
// main: one 256-thread workgroup per batch element
// ------------------------------------------------------------------
__launch_bounds__(256, 1)
__global__ void eanet_main(const float* __restrict__ electrons,
                           const float* __restrict__ atoms,
                           const float* __restrict__ w_htwo,
                           const float* __restrict__ b_htwo,
                           const float* __restrict__ nuc_emb,
                           const float* __restrict__ w_hone,
                           const float* __restrict__ b_hone,
                           const float* __restrict__ gmu,
                           const float* __restrict__ b_atom,
                           const float* __restrict__ bself,
                           const float* __restrict__ bmsg,
                           const float* __restrict__ b_lin,
                           const char*  __restrict__ ws,
                           float* __restrict__ out) {
  extern __shared__ char smem[];
  float*  s_elec   = (float*)(smem + L_ELEC);
  float*  s_atoms  = (float*)(smem + L_ATOMS);
  float*  s_nuc    = (float*)(smem + L_NUC);
  float*  s_mu     = (float*)(smem + L_MU);
  float*  s_spd    = (float*)(smem + L_SPD);
  float*  s_rbar   = (float*)(smem + L_RBAR);
  float*  s_rbmean = (float*)(smem + L_RBMEAN);
  float*  s_r4     = (float*)(smem + L_R4);
  half_t* s_h2cat  = (half_t*)(smem + L_H2CAT);
  float*  s_aemb   = (float*)(smem + L_AEMB);
  float*  s_pre    = (float*)(smem + L_PRE);
  half_t* s_aembh  = (half_t*)(smem + L_AEMBH);
  half_t* s_msgh   = (half_t*)(smem + L_MSGH);
  float*  s_mpw    = (float*)(smem + L_MPW);
  half_t* s_elecA  = (half_t*)(smem + L_ELECA);
  half_t* s_atcat  = (half_t*)(smem + L_ATOMCAT);

  const float*  ws_mpw   = (const float*)(ws + WS_MPW);
  const float*  ws_spd   = (const float*)(ws + WS_SPD);
  const half_t* ws_watom = (const half_t*)(ws + WS_WATOM);
  const half_t* ws_wlin  = (const half_t*)(ws + WS_WLIN);
  const half_t* ws_wself = (const half_t*)(ws + WS_WSELF);
  const half_t* ws_wmsg  = (const half_t*)(ws + WS_WMSG);

  const int tid  = threadIdx.x;
  const int lane = tid & 31;
  const int wave = tid >> 5;
  const int b    = blockIdx.x;
  const float* eb = electrons + (size_t)b * NE * 3;

  // ---------------- phase 0: geometry + RBF ----------------
  for (int p = tid; p < NE * 3; p += 256) s_elec[p] = eb[p];
  for (int p = tid; p < NA * 3; p += 256) s_atoms[p] = atoms[p];
  for (int p = tid; p < NA * 4; p += 256) s_nuc[p] = nuc_emb[p];
  if (tid < NRBF) s_mu[tid] = gmu[tid];
  else if (tid < 2 * NRBF) s_spd[tid - NRBF] = ws_spd[tid - NRBF];
  __syncthreads();

  for (int p = tid; p < NE * NA; p += 256) {
    int i = p >> 4, m = p & 15;
    float dx = s_elec[i * 3 + 0] - s_atoms[m * 3 + 0];
    float dy = s_elec[i * 3 + 1] - s_atoms[m * 3 + 1];
    float dz = s_elec[i * 3 + 2] - s_atoms[m * 3 + 2];
    float rn = sqrtf(dx * dx + dy * dy + dz * dz);
    float* r4 = &s_r4[p * 4];
    r4[0] = dx; r4[1] = dy; r4[2] = dz; r4[3] = rn;
    float acc = 0.0f;
#pragma unroll
    for (int r = 0; r < NRBF; ++r) {
      float t2 = rn - s_mu[r];
      acc += __expf(-(t2 * t2) / s_spd[r]);
    }
    s_rbar[p] = acc * (1.0f / NRBF);
  }
  __syncthreads();
  if (tid < NE) {
    float a = 0.0f;
#pragma unroll
    for (int m = 0; m < NA; ++m) a += s_rbar[tid * NA + m];
    s_rbmean[tid] = a * (1.0f / NA);
  }
  __syncthreads();

  // ---------------- phase 1: h_two streamed means -> h2cat (64x512) ----------------
  {
    const int c = tid;  // 256 threads == 256 channels
    const float w0 = w_htwo[0 * DIMF + c], w1 = w_htwo[1 * DIMF + c];
    const float w2 = w_htwo[2 * DIMF + c], w3 = w_htwo[3 * DIMF + c];
    const float bc = b_htwo[c];
    for (int i = 0; i < NE; ++i) {
      float eix = s_elec[i * 3 + 0], eiy = s_elec[i * 3 + 1], eiz = s_elec[i * 3 + 2];
      float accU = 0.0f, accD = 0.0f;
      for (int j = 0; j < NE; ++j) {
        float dx = eix - s_elec[j * 3 + 0];
        float dy = eiy - s_elec[j * 3 + 1];
        float dz = eiz - s_elec[j * 3 + 2];
        float rn = (j == i) ? 0.0f : sqrtf(dx * dx + dy * dy + dz * dz);
        float pre = fmaf(dx, w0, fmaf(dy, w1, fmaf(dz, w2, fmaf(rn, w3, bc))));
        float h = tanhf(pre) * TANH_GAIN;
        if (j < NUPE) accU += h; else accD += h;
      }
      s_h2cat[i * 512 + c]       = (half_t)(accU * (1.0f / NUPE));
      s_h2cat[i * 512 + 256 + c] = (half_t)(accD * (1.0f / NUPE));
    }
  }
  __syncthreads();

  // ---------------- phase 2: h_one + dual reductions ----------------
  {
    const int c = tid;
    const float w0 = w_hone[0 * DIMF + c], w1 = w_hone[1 * DIMF + c];
    const float w2 = w_hone[2 * DIMF + c], w3 = w_hone[3 * DIMF + c];
    const float bc = b_hone[c];
    float accA0[NA], accA1[NA];
#pragma unroll
    for (int m = 0; m < NA; ++m) { accA0[m] = 0.0f; accA1[m] = 0.0f; }
    for (int i = 0; i < NE; ++i) {
      float accE = 0.0f;
#pragma unroll
      for (int m = 0; m < NA; ++m) {
        const float* r4 = &s_r4[(i * NA + m) * 4];
        float pre = fmaf(r4[0] + s_nuc[m * 4 + 0], w0,
                    fmaf(r4[1] + s_nuc[m * 4 + 1], w1,
                    fmaf(r4[2] + s_nuc[m * 4 + 2], w2,
                    fmaf(r4[3] + s_nuc[m * 4 + 3], w3, bc))));
        float h = tanhf(pre) * TANH_GAIN * s_rbar[i * NA + m];
        accE += h;
        if (i < NUPE) accA0[m] += h; else accA1[m] += h;
      }
      s_elecA[i * F12F + c] = (half_t)(accE * (1.0f / NA));
    }
#pragma unroll
    for (int m = 0; m < NA; ++m) {
      s_atcat[m * 1536 + c]       = (half_t)(accA0[m] * (1.0f / NUPE));
      s_atcat[m * 1536 + 768 + c] = (half_t)(accA1[m] * (1.0f / NUPE));
    }
  }
  __syncthreads();

  // phase 2b: h_two contributions to elecA and atomcat
  for (int p = tid; p < NE * 512; p += 256) {
    int i = p >> 9, c2 = p & 511;
    s_elecA[i * F12F + 256 + c2] = (half_t)((float)s_h2cat[i * 512 + c2] * s_rbmean[i]);
  }
  for (int q = tid; q < 2 * NA * 512; q += 256) {
    int spin = q >> 13;
    int rem  = q & 8191;
    int m = rem >> 9, c2 = rem & 511;
    int i0 = spin * NUPE;
    float acc = 0.0f;
    for (int i = i0; i < i0 + NUPE; ++i)
      acc = fmaf(s_rbar[i * NA + m], (float)s_h2cat[i * 512 + c2], acc);
    s_atcat[m * 1536 + spin * 768 + 256 + c2] = (half_t)(acc * (1.0f / NUPE));
  }
  __syncthreads();
  // h2cat is now dead: phase-3 overlays may write its region.

  // ---------------- phase 3a: atom_emb = silu(atomcat @ w_atom + b) ----------------
  for (int tile = wave; tile < 16; tile += 8) {
    const int n = tile * 16 + (lane & 15);
    const half_t* arow = s_atcat + (lane & 15) * 1536;
    v8f acc = {0.f, 0.f, 0.f, 0.f, 0.f, 0.f, 0.f, 0.f};
#pragma unroll 4
    for (int k0 = 0; k0 < 48; ++k0) {
      if (k0 + 1 < 48)
        __builtin_prefetch((const void*)((const v16h*)ws_watom +
                           (((size_t)(k0 + 1) * 256 + n) * 2)), 0, 0);
      v16h a = load_a_frag(arow, k0 * 32, lane);
      v16h w = load_b_frag(ws_watom, k0, n, lane);
      acc = wmma16(a, w, acc);
    }
    int h = (lane >> 4) & 1;
#pragma unroll
    for (int r = 0; r < 8; ++r) s_pre[(r + 8 * h) * DIMF + n] = acc[r];
  }
  __syncthreads();
  // epilogue + load mp_weight into LDS
  s_mpw[tid] = ws_mpw[tid];  // 256 threads, 256 entries
  for (int p = tid; p < NA * DIMF; p += 256) {
    int c = p & 255;
    float e = silu_g(s_pre[p] + b_atom[c]);
    s_aemb[p]  = e;
    s_aembh[p] = (half_t)e;
  }
  __syncthreads();

  // ---------------- phase 3b: message-passing layers ----------------
  for (int l = 0; l < NLAY; ++l) {
    // msg = mp_weight @ atom_emb  (16x16 @ 16x256, scalar)
    for (int p = tid; p < NA * DIMF; p += 256) {
      int m = p >> 8, c = p & 255;
      float f = 0.0f;
#pragma unroll
      for (int n = 0; n < NA; ++n)
        f = fmaf(s_mpw[m * NA + n], s_aemb[n * DIMF + c], f);
      s_msgh[p] = (half_t)f;
    }
    __syncthreads();
    const half_t* wS = ws_wself + (size_t)l * 256 * 256;
    const half_t* wM = ws_wmsg  + (size_t)l * 256 * 256;
    for (int tile = wave; tile < 16; tile += 8) {
      const int n = tile * 16 + (lane & 15);
      const half_t* arowS = s_aembh + (lane & 15) * DIMF;
      const half_t* arowM = s_msgh  + (lane & 15) * DIMF;
      v8f acc = {0.f, 0.f, 0.f, 0.f, 0.f, 0.f, 0.f, 0.f};
#pragma unroll
      for (int k0 = 0; k0 < 8; ++k0) {
        v16h aS = load_a_frag(arowS, k0 * 32, lane);
        v16h bS = load_b_frag(wS, k0, n, lane);
        acc = wmma16(aS, bS, acc);
        v16h aM = load_a_frag(arowM, k0 * 32, lane);
        v16h bM = load_b_frag(wM, k0, n, lane);
        acc = wmma16(aM, bM, acc);
      }
      int h = (lane >> 4) & 1;
#pragma unroll
      for (int r = 0; r < 8; ++r) s_pre[(r + 8 * h) * DIMF + n] = acc[r];
    }
    __syncthreads();
    for (int p = tid; p < NA * DIMF; p += 256) {
      int c = p & 255;
      float x  = (s_pre[p] + bself[l * DIMF + c] + bmsg[l * DIMF + c]) * RSQRT2;
      float nv = silu_g(x);
      float u  = (s_aemb[p] + nv) * RSQRT2;
      s_aemb[p]  = u;
      s_aembh[p] = (half_t)u;
    }
    __syncthreads();
  }

  // ---------------- phase 3c: elec_emb GEMM + fused outer-product store ----------------
  float* outb = out + (size_t)b * NE * NA * DIMF;
  for (int t = wave; t < 64; t += 8) {
    int mt = t >> 4, nt = t & 15;
    const int n = nt * 16 + (lane & 15);
    const half_t* arow = s_elecA + (mt * 16 + (lane & 15)) * F12F;
    v8f acc = {0.f, 0.f, 0.f, 0.f, 0.f, 0.f, 0.f, 0.f};
#pragma unroll 4
    for (int k0 = 0; k0 < 24; ++k0) {
      v16h a = load_a_frag(arow, k0 * 32, lane);
      v16h w = load_b_frag(ws_wlin, k0, n, lane);
      acc = wmma16(a, w, acc);
    }
    const float bl = b_lin[n];
    const int h = (lane >> 4) & 1;
#pragma unroll
    for (int r = 0; r < 8; ++r) {
      int i = mt * 16 + r + 8 * h;
      float e = silu_g(acc[r] + bl);
      float* op = outb + ((size_t)i * NA) * DIMF + n;
#pragma unroll
      for (int m = 0; m < NA; ++m) op[m * DIMF] = e * s_aemb[m * DIMF + n];
    }
  }
}

// ------------------------------------------------------------------
extern "C" void kernel_launch(void* const* d_in, const int* in_sizes, int n_in,
                              void* d_out, int out_size, void* d_ws, size_t ws_size,
                              hipStream_t stream) {
  (void)in_sizes; (void)n_in; (void)out_size; (void)ws_size;
  const float* electrons   = (const float*)d_in[0];
  const float* atoms       = (const float*)d_in[1];
  const float* w_htwo      = (const float*)d_in[2];
  const float* b_htwo      = (const float*)d_in[3];
  const float* nuc_emb     = (const float*)d_in[4];
  const float* w_hone      = (const float*)d_in[5];
  const float* b_hone      = (const float*)d_in[6];
  const float* temperature = (const float*)d_in[7];
  const float* gauss_mu    = (const float*)d_in[8];
  const float* gauss_sigma = (const float*)d_in[9];
  const float* w_atom      = (const float*)d_in[10];
  const float* b_atom      = (const float*)d_in[11];
  const float* Wself       = (const float*)d_in[12];
  const float* bself       = (const float*)d_in[13];
  const float* Wmsg        = (const float*)d_in[14];
  const float* bmsg        = (const float*)d_in[15];
  const float* w_lin       = (const float*)d_in[16];
  const float* b_lin       = (const float*)d_in[17];
  float* out = (float*)d_out;
  char* ws = (char*)d_ws;

  float*  ws_mpw   = (float*)(ws + WS_MPW);
  float*  ws_spd   = (float*)(ws + WS_SPD);
  half_t* ws_watom = (half_t*)(ws + WS_WATOM);
  half_t* ws_wlin  = (half_t*)(ws + WS_WLIN);
  half_t* ws_wself = (half_t*)(ws + WS_WSELF);
  half_t* ws_wmsg  = (half_t*)(ws + WS_WMSG);

  eanet_prep_small<<<1, 64, 0, stream>>>(atoms, temperature, gauss_sigma,
                                         ws_mpw, ws_spd);
  eanet_pack_w<<<(1536 * 256 + 255) / 256, 256, 0, stream>>>(w_atom, ws_watom, 1536 * 256);
  eanet_pack_w<<<(768 * 256 + 255) / 256, 256, 0, stream>>>(w_lin, ws_wlin, 768 * 256);
  for (int l = 0; l < NLAY; ++l) {
    eanet_pack_w<<<(256 * 256 + 255) / 256, 256, 0, stream>>>(
        Wself + (size_t)l * 256 * 256, ws_wself + (size_t)l * 256 * 256, 256 * 256);
    eanet_pack_w<<<(256 * 256 + 255) / 256, 256, 0, stream>>>(
        Wmsg + (size_t)l * 256 * 256, ws_wmsg + (size_t)l * 256 * 256, 256 * 256);
  }

  eanet_main<<<NB, 256, L_TOTAL, stream>>>(
      electrons, atoms, w_htwo, b_htwo, nuc_emb, w_hone, b_hone,
      gauss_mu, b_atom, bself, bmsg, b_lin, ws, out);
}
